// RecognitionODEGRU_37649683317033
// MI455X (gfx1250) — matrix operand order repaired and checked by reading, not compiled, not run on hardware
//
#include <hip/hip_runtime.h>

// ---------------------------------------------------------------------------
// Problem constants (from reference): B=512, T=8, D=512, H=512, LAT=64, NSTEPS=4
// ---------------------------------------------------------------------------
#define BB 512
#define TT 8
#define DD 512
#define HH 512

typedef __attribute__((ext_vector_type(16))) _Float16 v16h;
typedef __attribute__((ext_vector_type(8)))  float    v8f;
typedef __attribute__((ext_vector_type(4)))  unsigned int u32x4;

union Frag { v16h h; u32x4 q[2]; };

#define TM 128
#define KB 32
#define LDSROW 40   // 32 halves + 8 pad = 80 bytes per row (16B aligned)

// ---------------------------------------------------------------------------
// Fused WMMA GEMM:  C[M x N] = act( (A0 + alpha*dt[m]*A1) @ W^T + bias
//                                   + tw[n]*(t0[m] + tshift*dt[m]) )
// W is (N x K) f16 row-major.  Optionally also Cacc = (init?0:Cacc) + accW*C.
// Block: 256 threads, C tile 128 x TNT.  Double-buffered LDS; B operand is
// fetched with GLOBAL_LOAD_ASYNC_TO_LDS_B128 (ASYNCcnt), A is staged through
// VGPRs (fp32->f16 conversion + fused RK4 combine) one chunk ahead.
// ---------------------------------------------------------------------------
template <int TNT>
__global__ void __launch_bounds__(256)
gemm_f16wmma(const float* __restrict__ A0, const float* __restrict__ A1,
             const float* __restrict__ dtv, float alpha,
             const _Float16* __restrict__ W, const float* __restrict__ bias,
             const float* __restrict__ tw, const float* __restrict__ t0v, float tshift,
             float* __restrict__ C, float* __restrict__ Cacc, float accW, int accInit,
             int N, int K, int lda, int actSwish)
{
  constexpr int WN    = TNT / 64;   // waves along N (4 or 2)
  constexpr int WM    = 8 / WN;     // waves along M (2 or 4)
  constexpr int MT    = 8 / WM;     // 16x16 M-tiles per wave (4 or 2)
  constexpr int ASZ   = TM * LDSROW;
  constexpr int BSZ   = TNT * LDSROW;
  constexpr int BPASS = TNT / 64;   // async b128 issues per thread per chunk

  __shared__ _Float16 Asmem[2 * ASZ];
  __shared__ _Float16 Bsmem[2 * BSZ];

  const int tid  = threadIdx.x;
  const int lane = tid & 31;
  const int wid  = tid >> 5;
  const int wn   = wid % WN;
  const int wm   = wid / WN;
  const int m0   = blockIdx.y * TM;
  const int n0   = blockIdx.x * TNT;
  const int lh   = lane & 15;
  const int hi   = lane >> 4;
  const bool useA1 = (A1 != nullptr);

  v8f acc[MT][4];
#pragma unroll
  for (int i = 0; i < MT; ++i)
#pragma unroll
    for (int j = 0; j < 4; ++j)
#pragma unroll
      for (int e = 0; e < 8; ++e) acc[i][j][e] = 0.f;

  float2 areg[8];

  auto loadA = [&](int kb) {
    if (useA1) {
#pragma unroll
      for (int it = 0; it < 8; ++it) {
        int linear = it * 256 + tid;            // 0..2047, 2 elems each
        int m = linear >> 4, kk = (linear & 15) << 1;
        float2 v = *(const float2*)&A0[(size_t)(m0 + m) * lda + kb + kk];
        float2 w = *(const float2*)&A1[(size_t)(m0 + m) * lda + kb + kk];
        float sc = alpha * dtv[m0 + m];
        v.x += sc * w.x; v.y += sc * w.y;
        areg[it] = v;
      }
    } else {
#pragma unroll
      for (int it = 0; it < 8; ++it) {
        int linear = it * 256 + tid;
        int m = linear >> 4, kk = (linear & 15) << 1;
        areg[it] = *(const float2*)&A0[(size_t)(m0 + m) * lda + kb + kk];
      }
    }
  };
  auto storeA = [&](_Float16* dst) {
#pragma unroll
    for (int it = 0; it < 8; ++it) {
      int linear = it * 256 + tid;
      int m = linear >> 4, kk = (linear & 15) << 1;
      dst[m * LDSROW + kk]     = (_Float16)areg[it].x;
      dst[m * LDSROW + kk + 1] = (_Float16)areg[it].y;
    }
  };
  // B tile: per-lane async DMA global -> LDS (16B each), tracked by ASYNCcnt.
  auto loadB_async = [&](int kb, _Float16* dst) {
#pragma unroll
    for (int it = 0; it < BPASS; ++it) {
      int linear = it * 256 + tid;              // TNT*4 chunks of 8 halves
      int n = linear >> 2, ko = (linear & 3) << 3;
      unsigned ldsoff = (unsigned)(unsigned long long)&dst[n * LDSROW + ko];
      unsigned long long g =
          (unsigned long long)(size_t)&W[(size_t)(n0 + n) * K + kb + ko];
      asm volatile("global_load_async_to_lds_b128 %0, %1, off"
                   :: "v"(ldsoff), "v"(g) : "memory");
    }
  };

  // ---- software pipeline prologue ----
  loadB_async(0, Bsmem);
  loadA(0);
  storeA(Asmem);
  asm volatile("s_wait_asynccnt 0x0" ::: "memory");
  __syncthreads();

  int c = 0;
  for (int kb = 0; kb < K; kb += KB) {
    _Float16* Acur = Asmem + c * ASZ;
    _Float16* Bcur = Bsmem + c * BSZ;
    _Float16* Anxt = Asmem + (1 - c) * ASZ;
    _Float16* Bnxt = Bsmem + (1 - c) * BSZ;
    const bool notlast = (kb + KB < K);

    if (notlast) { loadB_async(kb + KB, Bnxt); loadA(kb + KB); }

    // ---- fragments per documented 16-bit A(16x32)/B(32x16) layouts ----
    Frag fa[MT], fb[4];
#pragma unroll
    for (int i = 0; i < MT; ++i) {
      int arow = wm * (MT * 16) + i * 16 + lh;
      const _Float16* ap = &Acur[arow * LDSROW + hi * 8];
      fa[i].q[0] = *(const u32x4*)ap;         // K = hi*8 .. +7
      fa[i].q[1] = *(const u32x4*)(ap + 16);  // K = hi*8+16 .. +23
    }
#pragma unroll
    for (int j = 0; j < 4; ++j) {
      int brow = wn * 64 + j * 16 + lh;
      const _Float16* bp = &Bcur[brow * LDSROW + hi * 8];
      fb[j].q[0] = *(const u32x4*)bp;         // K = hi*8 .. +7
      fb[j].q[1] = *(const u32x4*)(bp + 16);  // K = hi*8+16 .. +23
    }
#pragma unroll
    for (int i = 0; i < MT; ++i)
#pragma unroll
      for (int j = 0; j < 4; ++j)
        acc[i][j] = __builtin_amdgcn_wmma_f32_16x16x32_f16(
            false, fa[i].h, false, fb[j].h, (short)0, acc[i][j], false, false);

    if (notlast) storeA(Anxt);
    asm volatile("s_wait_asynccnt 0x0" ::: "memory");
    __syncthreads();
    c ^= 1;
  }

  // ---- epilogue: bias + t-column + swish + dual write ----
#pragma unroll
  for (int i = 0; i < MT; ++i) {
#pragma unroll
    for (int j = 0; j < 4; ++j) {
      int nn = n0 + wn * 64 + j * 16 + lh;
      float bv  = bias ? bias[nn] : 0.f;
      float twv = tw ? tw[nn] : 0.f;
#pragma unroll
      for (int v = 0; v < 8; ++v) {
        int mm = m0 + wm * (MT * 16) + i * 16 + v + hi * 8;
        float val = acc[i][j][v] + bv;
        if (tw) val += twv * (t0v[mm] + tshift * dtv[mm]);
        if (actSwish) val = val * (1.f / (1.f + __expf(-val)));
        size_t idx = (size_t)mm * N + nn;
        C[idx] = val;
        if (Cacc) {
          float p = accInit ? 0.f : Cacc[idx];
          Cacc[idx] = p + accW * val;
        }
      }
    }
  }
}

// ---------------------------------------------------------------------------
// Small elementwise kernels
// ---------------------------------------------------------------------------
__global__ void convert_f16(const float* __restrict__ s, _Float16* __restrict__ d, int n) {
  int i = blockIdx.x * 256 + threadIdx.x;
  if (i < n) d[i] = (_Float16)s[i];
}
__global__ void w1split(const float* __restrict__ W1, _Float16* __restrict__ w1h,
                        float* __restrict__ tw1) {
  int i = blockIdx.x * 256 + threadIdx.x;
  if (i < 1024 * 512) { int n = i >> 9, k = i & 511; w1h[i] = (_Float16)W1[n * 513 + k]; }
  if (i < 1024) tw1[i] = W1[i * 513 + 512];
}
__global__ void zero_f32(float* __restrict__ p, int n) {
  int i = blockIdx.x * 256 + threadIdx.x;
  if (i < n) p[i] = 0.f;
}
__global__ void copy_f32(const float* __restrict__ s, float* __restrict__ d, int n) {
  int i = blockIdx.x * 256 + threadIdx.x;
  if (i < n) d[i] = s[i];
}
__global__ void tstep_kernel(const float* __restrict__ ts, int t,
                             float* __restrict__ t0, float* __restrict__ dtb) {
  int i = blockIdx.x * 256 + threadIdx.x;
  if (i < BB) {
    float t1 = ts[i * TT + t];
    float tp = (t == 0) ? ts[i * TT] : ts[i * TT + t - 1];
    t0[i] = tp;
    dtb[i] = (t1 - tp) * 0.25f;   // /NSTEPS
  }
}
__global__ void h_update(float* __restrict__ h, const float* __restrict__ kacc,
                         const float* __restrict__ dtb) {
  int i = blockIdx.x * 256 + threadIdx.x;
  if (i < BB * HH) { int m = i >> 9; h[i] += dtb[m] * (1.f / 6.f) * kacc[i]; }
}
__global__ void gru_gates(const float* __restrict__ gi, const float* __restrict__ gh,
                          float* __restrict__ h) {
  int i = blockIdx.x * 256 + threadIdx.x;
  if (i < BB * HH) {
    int m = i >> 9, j = i & 511;
    const float* gim = gi + (size_t)m * 1536;
    const float* ghm = gh + (size_t)m * 1536;
    float ir = gim[j], iz = gim[512 + j], in = gim[1024 + j];
    float hr = ghm[j], hz = ghm[512 + j], hn = ghm[1024 + j];
    float r  = 1.f / (1.f + __expf(-(ir + hr)));
    float zg = 1.f / (1.f + __expf(-(iz + hz)));
    float n  = tanhf(in + r * hn);
    h[i] = (1.f - zg) * n + zg * h[i];
  }
}

// ---------------------------------------------------------------------------
static inline void gemm(hipStream_t s, const float* A0, const float* A1, const float* dtv,
                        float alpha, const _Float16* W, const float* bias, const float* tw,
                        const float* t0v, float tshift, float* C, float* Cacc,
                        float accW, int accInit, int M, int N, int K, int lda, int act) {
  if (N % 256 == 0) {
    dim3 g(N / 256, M / 128), b(256);
    gemm_f16wmma<256><<<g, b, 0, s>>>(A0, A1, dtv, alpha, W, bias, tw, t0v, tshift,
                                      C, Cacc, accW, accInit, N, K, lda, act);
  } else {
    dim3 g(N / 128, M / 128), b(256);
    gemm_f16wmma<128><<<g, b, 0, s>>>(A0, A1, dtv, alpha, W, bias, tw, t0v, tshift,
                                      C, Cacc, accW, accInit, N, K, lda, act);
  }
}
#define EWG(n) dim3(((n) + 255) / 256), dim3(256), 0, stream

extern "C" void kernel_launch(void* const* d_in, const int* in_sizes, int n_in,
                              void* d_out, int out_size, void* d_ws, size_t ws_size,
                              hipStream_t stream) {
  const float* x    = (const float*)d_in[0];
  const float* ts   = (const float*)d_in[1];
  const float* Wih  = (const float*)d_in[2];
  const float* Whh  = (const float*)d_in[3];
  const float* bih  = (const float*)d_in[4];
  const float* bhh  = (const float*)d_in[5];
  const float* Wout = (const float*)d_in[6];
  const float* bout = (const float*)d_in[7];
  const float* W1   = (const float*)d_in[8];
  const float* b1   = (const float*)d_in[9];
  const float* W2   = (const float*)d_in[10];
  const float* b2   = (const float*)d_in[11];
  const float* W3   = (const float*)d_in[12];
  const float* b3   = (const float*)d_in[13];
  float* outp = (float*)d_out;

  // Workspace carve (floats). ODE buffers and GRU buffers alias region R.
  float* ws   = (float*)d_ws;
  float* hbuf = ws;                       // 512*512
  float* R    = ws + 262144;              // 1572864 floats
  float* kcur = R;                        // 512*512
  float* kacc = R + 262144;               // 512*512
  float* z1   = R + 524288;               // 512*1024
  float* z2   = R + 1048576;              // 512*1024
  float* gib  = R;                        // 512*1536 (aliases kcur/kacc/z1 half)
  float* ghb  = R + 786432;               // 512*1536
  float* t0b  = ws + 1835008;             // 512
  float* dtb  = ws + 1835520;             // 512
  float* tw1  = ws + 1836032;             // 1024
  _Float16* f16  = (_Float16*)(ws + 1837056);
  _Float16* wih16  = f16;                 // 1536*512
  _Float16* whh16  = wih16 + 786432;      // 1536*512
  _Float16* w1h16  = whh16 + 786432;      // 1024*512
  _Float16* w2_16  = w1h16 + 524288;      // 1024*1024
  _Float16* w3_16  = w2_16 + 1048576;     // 512*1024
  _Float16* wout16 = w3_16 + 524288;      // 128*512

  // --- weight pre-conversion to f16 (deterministic, every call) ---
  convert_f16<<<EWG(786432)>>>(Wih, wih16, 786432);
  convert_f16<<<EWG(786432)>>>(Whh, whh16, 786432);
  convert_f16<<<EWG(1048576)>>>(W2, w2_16, 1048576);
  convert_f16<<<EWG(524288)>>>(W3, w3_16, 524288);
  convert_f16<<<EWG(65536)>>>(Wout, wout16, 65536);
  w1split<<<EWG(1024 * 512)>>>(W1, w1h16, tw1);

  zero_f32<<<EWG(BB * HH)>>>(hbuf, BB * HH);   // h0 = 0

  for (int t = 0; t < TT; ++t) {
    tstep_kernel<<<EWG(BB)>>>(ts, t, t0b, dtb);

    // ---- ODE integrate: NSTEPS RK4 steps ----
    for (int j = 0; j < 4; ++j) {
      float js = (float)j;
      const float alphas[4] = {0.f, 0.5f, 0.5f, 1.f};
      const float shifts[4] = {0.f, 0.5f, 0.5f, 1.f};
      const float accws[4]  = {1.f, 2.f, 2.f, 1.f};
      for (int e = 0; e < 4; ++e) {
        const float* A1 = (e == 0) ? nullptr : kcur;
        // layer 1: z1 = swish((h + a*dt*k) @ W1h^T + t*w1t + b1)
        gemm(stream, hbuf, A1, dtb, alphas[e], w1h16, b1, tw1, t0b, js + shifts[e],
             z1, nullptr, 0.f, 0, BB, 1024, 512, 512, 1);
        // layer 2: z2 = swish(z1 @ W2^T + b2)
        gemm(stream, z1, nullptr, nullptr, 0.f, w2_16, b2, nullptr, nullptr, 0.f,
             z2, nullptr, 0.f, 0, BB, 1024, 1024, 1024, 1);
        // layer 3: k_e = z2 @ W3^T + b3 ; kacc (+)= w_e * k_e
        gemm(stream, z2, nullptr, nullptr, 0.f, w3_16, b3, nullptr, nullptr, 0.f,
             kcur, kacc, accws[e], (e == 0) ? 1 : 0, BB, 512, 1024, 1024, 0);
      }
      h_update<<<EWG(BB * HH)>>>(hbuf, kacc, dtb);  // h += dt/6 * kacc
    }

    // ---- GRU cell ----
    gemm(stream, x + (size_t)t * DD, nullptr, nullptr, 0.f, wih16, bih, nullptr, nullptr,
         0.f, gib, nullptr, 0.f, 0, BB, 1536, 512, TT * DD, 0);
    gemm(stream, hbuf, nullptr, nullptr, 0.f, whh16, bhh, nullptr, nullptr,
         0.f, ghb, nullptr, 0.f, 0, BB, 1536, 512, 512, 0);
    gru_gates<<<EWG(BB * HH)>>>(gib, ghb, hbuf);

    // ---- output projection (only last step's out is returned) ----
    if (t == TT - 1)
      gemm(stream, hbuf, nullptr, nullptr, 0.f, wout16, bout, nullptr, nullptr,
           0.f, outp, nullptr, 0.f, 0, BB, 128, 512, 512, 0);
  }

  // d_out layout: out (512*128) | h (512*512) | c (512*512, zeros)
  copy_f32<<<EWG(BB * HH)>>>(hbuf, outp + 65536, BB * HH);
  zero_f32<<<EWG(BB * HH)>>>(outp + 65536 + 262144, BB * HH);
}
